// GraphAttention_72808285602171
// MI455X (gfx1250) — compile-verified, matrix operand order checked by reading, not run on hardware
//
#include <hip/hip_runtime.h>
#include <hip/hip_bf16.h>
#include <math.h>

typedef __attribute__((ext_vector_type(16))) _Float16 v16h;
typedef __attribute__((ext_vector_type(8)))  float    v8f;

#define NN     32768
#define NE     262144
#define CSC    32
#define CVC    16
#define HH     4
#define NBANK  16
#define OSS    48      // CS + CV
#define WN     112
#define FEAT   80
#define FLATC  160

#define INV_SQRT3 0.5773502691896258f
#define INV_SQRT2 0.7071067811865476f

// ---------------------------------------------------------------------------
// WMMA helpers (gfx1250: V_WMMA_F32_16X16X32_F16, wave32)
// ---------------------------------------------------------------------------
__device__ __forceinline__ v8f wmma16(v16h a, v16h b, v8f c) {
  return __builtin_amdgcn_wmma_f32_16x16x32_f16(false, a, false, b, (short)0, c,
                                                false, false);
}

// A-matrix fragment from LDS tile (row-major [rows][rowHalfs] halfs).
// ISA 16-bit A layout: lane l -> row M = row0 + (l&15), half h = l>>4;
// element i -> K = 32*kstep + 16*(i>>3) + 2*((i>>1)&3) + (i&1) + 8*h.
// Consecutive element pairs are consecutive K -> b32/b128-mergeable loads.
__device__ __forceinline__ v16h frag_a(const _Float16* base, int row0, int kstep,
                                       int rowHalfs) {
  union { v16h v; unsigned u[8]; } f;
  const int lane = threadIdx.x & 31;
  const int m  = row0 + (lane & 15);
  const int hh = lane >> 4;
  const unsigned* u32 = (const unsigned*)base;
  const int rb = m * (rowHalfs >> 1);
#pragma unroll
  for (int p = 0; p < 8; ++p) {
    int k0 = (kstep << 5) + ((p >> 2) << 4) + ((p & 3) << 1) + (hh << 3);
    f.u[p] = u32[rb + (k0 >> 1)];
  }
  return f.v;
}

// Stage a row-major f32 weight [rows][ncols] into fragment-ordered f16 LDS:
// dst[((kstep*ntiles + ntile)*32 + lane)*16 + i], zero-padding K rows >= rows.
// Each B fragment then loads as one contiguous, 32B-aligned v16h.
__device__ __forceinline__ void stage_b_swizzled(_Float16* dst,
                                                 const float* src, int rows,
                                                 int ncols, int ntiles,
                                                 int ksteps) {
  const int total = ksteps * ntiles * 512;
  for (int idx = threadIdx.x; idx < total; idx += blockDim.x) {
    int i    = idx & 15;
    int lane = (idx >> 4) & 31;
    int grp  = idx >> 9;            // = kstep*ntiles + ntile
    int nt   = grp % ntiles;
    int ks   = grp / ntiles;
    int n = nt * 16 + (lane & 15);
    int k = ks * 32 + ((i >> 3) << 4) + (((i >> 1) & 3) << 1) + (i & 1) +
            ((lane >> 4) << 3);
    dst[idx] = (k < rows) ? (_Float16)src[k * ncols + n] : (_Float16)0.f;
  }
}

__device__ __forceinline__ v16h frag_b_sw(const _Float16* base, int ntiles,
                                          int ntile, int kstep) {
  const int lane = threadIdx.x & 31;
  return *(const v16h*)(base + (((kstep * ntiles + ntile) * 32 + lane) << 4));
}

__device__ __forceinline__ void atomicMaxF(float* addr, float val) {
  float old = *((volatile float*)addr);
  while (val > old) {
    int assumed = __float_as_int(old);
    int prev = atomicCAS((int*)addr, assumed, __float_as_int(val));
    if (prev == assumed) break;
    old = __int_as_float(prev);
  }
}

// ---------------------------------------------------------------------------
// K0: init workspace
// ---------------------------------------------------------------------------
__global__ void init_kernel(float* mbuf, float* denom, float* agg) {
  int i = blockIdx.x * blockDim.x + threadIdx.x;
  if (i < NN * HH) { mbuf[i] = -3.0e38f; denom[i] = 0.f; }
  if (i < NN * FLATC) agg[i] = 0.f;
}

// ---------------------------------------------------------------------------
// K1: per-node O(3) layernorm + src/dst linears
// ---------------------------------------------------------------------------
__global__ void __launch_bounds__(256) node_kernel(
    const float* __restrict__ node,
    const float* __restrict__ Wsrc_s, const float* __restrict__ Wsrc_v,
    const float* __restrict__ Wdst_s, const float* __restrict__ Wdst_v,
    float* __restrict__ ssb, float* __restrict__ dsb,
    float* __restrict__ svb, float* __restrict__ dvb) {
  __shared__ float sWss[CSC * CSC], sWds[CSC * CSC];
  __shared__ float sWsv[CVC * CVC], sWdv[CVC * CVC];
  for (int i = threadIdx.x; i < CSC * CSC; i += blockDim.x) {
    sWss[i] = Wsrc_s[i]; sWds[i] = Wdst_s[i];
  }
  for (int i = threadIdx.x; i < CVC * CVC; i += blockDim.x) {
    sWsv[i] = Wsrc_v[i]; sWdv[i] = Wdst_v[i];
  }
  __syncthreads();
  int n = blockIdx.x * blockDim.x + threadIdx.x;
  if (n >= NN) return;
  const float* row = node + (size_t)n * FEAT;
  float s[CSC];
  float mean = 0.f, msq = 0.f;
#pragma unroll
  for (int c = 0; c < CSC; ++c) { s[c] = row[c]; mean += s[c]; msq += s[c] * s[c]; }
  mean *= (1.f / CSC);
  float var = msq * (1.f / CSC) - mean * mean;
  float inv = rsqrtf(fmaxf(var, 0.f) + 1e-5f);
#pragma unroll
  for (int c = 0; c < CSC; ++c) s[c] = (s[c] - mean) * inv;

  float v[CVC][3]; float n2 = 0.f;
#pragma unroll
  for (int c = 0; c < CVC; ++c)
#pragma unroll
    for (int d = 0; d < 3; ++d) {
      v[c][d] = row[CSC + c * 3 + d];
      n2 += v[c][d] * v[c][d];
    }
  n2 *= (1.f / CVC);
  float invv = rsqrtf(n2 + 1e-5f);
#pragma unroll
  for (int c = 0; c < CVC; ++c)
#pragma unroll
    for (int d = 0; d < 3; ++d) v[c][d] *= invv;

#pragma unroll
  for (int j = 0; j < CSC; ++j) {
    float a = 0.f, b = 0.f;
#pragma unroll
    for (int c = 0; c < CSC; ++c) {
      a += s[c] * sWss[c * CSC + j];
      b += s[c] * sWds[c * CSC + j];
    }
    ssb[(size_t)n * CSC + j] = a;
    dsb[(size_t)n * CSC + j] = b;
  }
#pragma unroll
  for (int j = 0; j < CVC; ++j)
#pragma unroll
    for (int d = 0; d < 3; ++d) {
      float a = 0.f, b = 0.f;
#pragma unroll
      for (int c = 0; c < CVC; ++c) {
        a += v[c][d] * sWsv[c * CVC + j];
        b += v[c][d] * sWdv[c * CVC + j];
      }
      svb[(size_t)n * 48 + j * 3 + d] = a;
      dvb[(size_t)n * 48 + j * 3 + d] = b;
    }
}

// ---------------------------------------------------------------------------
// K2: attention logits.  Per wave: 32 edges.
//     rbf@W_rbf (needed slices) via 6 WMMA, ms_@Wa via 16 WMMA,
//     head reduction with shfl_xor, segment max via CAS.
// ---------------------------------------------------------------------------
__global__ void __launch_bounds__(128) attn_kernel(
    const float* __restrict__ ssb, const float* __restrict__ dsb,
    const float* __restrict__ svb, const float* __restrict__ dvb,
    const float* __restrict__ rbf, const float* __restrict__ rsh,
    const float* __restrict__ W_rbf, const float* __restrict__ Wa,
    const float* __restrict__ alpha_dot,
    const int* __restrict__ srcI, const int* __restrict__ dstI,
    float* __restrict__ logit, float* __restrict__ mbuf) {
  __shared__ __align__(32) _Float16 sWaSw[2 * 4 * 512];   // swizzled Wa
  __shared__ __align__(32) _Float16 sWr3[3 * 512];        // W_rbf slots {0,1,5}
  __shared__ __align__(32) _Float16 sBigA[4][32 * 64];    // rbf tile, then ms_
  __shared__ __align__(32) _Float16 sWe[4][32 * 48];      // radial coeffs
  __shared__ int sDstW[4][32];

  stage_b_swizzled(sWaSw, Wa, OSS, 64, 4, 2);
  for (int idx = threadIdx.x; idx < 3 * 512; idx += blockDim.x) {
    int i = idx & 15, lane = (idx >> 4) & 31, s = idx >> 9;
    int nt = (s == 2) ? 5 : s;                 // cols 0-15,16-31,80-95
    int n = nt * 16 + (lane & 15);
    int k = ((i >> 3) << 4) + (((i >> 1) & 3) << 1) + (i & 1) +
            ((lane >> 4) << 3);
    sWr3[idx] = (k < NBANK) ? (_Float16)W_rbf[k * WN + n] : (_Float16)0.f;
  }
  __syncthreads();

  const int wave = threadIdx.x >> 5;
  const int lane = threadIdx.x & 31;
  const int e0 = (blockIdx.x * 4 + wave) * 32;
  const int e  = e0 + lane;
  _Float16* RbfA = sBigA[wave];   // 32x32 halfs (first half of buffer)
  _Float16* Ams  = sBigA[wave];   // 32x64 halfs (after rbf GEMM retires)
  _Float16* We   = sWe[wave];
  int* Dst = sDstW[wave];

  const int sI = srcI[e], dI = dstI[e];
  Dst[lane] = dI;
#pragma unroll
  for (int b = 0; b < NBANK; ++b)
    RbfA[lane * 32 + b] = (_Float16)rbf[(size_t)e * NBANK + b];
#pragma unroll
  for (int b = NBANK; b < 32; ++b) RbfA[lane * 32 + b] = (_Float16)0.f;

  const float y0 = rsh[(size_t)e * 4 + 0];
  float y1[3];
#pragma unroll
  for (int d = 0; d < 3; ++d) y1[d] = rsh[(size_t)e * 4 + 1 + d];

  float ms0[CSC];
#pragma unroll
  for (int c = 0; c < CSC; ++c)
    ms0[c] = ssb[(size_t)sI * CSC + c] + dsb[(size_t)dI * CSC + c];
  float dts[CVC];
#pragma unroll
  for (int c = 0; c < CVC; ++c) {
    float dt = 0.f;
#pragma unroll
    for (int d = 0; d < 3; ++d)
      dt += (svb[(size_t)sI * 48 + c * 3 + d] + dvb[(size_t)dI * 48 + c * 3 + d]) * y1[d];
    dts[c] = dt * INV_SQRT3;
  }
  __syncthreads();

  // radial weights: rbf(32x32) @ W_rbf-slices -> We  (6 WMMA)
#pragma unroll
  for (int t = 0; t < 2; ++t) {
    v16h a = frag_a(RbfA, t * 16, 0, 32);
#pragma unroll
    for (int s = 0; s < 3; ++s) {
      v8f c = {};
      c = wmma16(a, frag_b_sw(sWr3, 3, s, 0), c);
      const int n = s * 16 + (lane & 15);
      const int hh = lane >> 4;
#pragma unroll
      for (int r = 0; r < 8; ++r)
        We[(t * 16 + r + hh * 8) * 48 + n] = (_Float16)c[r];
    }
  }
  __syncthreads();

  // ms_ = [w_ss * ms0 * y0 | w_vd * (mv0.y1)/sqrt3], K padded to 64
#pragma unroll
  for (int c = 0; c < CSC; ++c)
    Ams[lane * 64 + c] = (_Float16)((float)We[lane * 48 + c] * ms0[c] * y0);
#pragma unroll
  for (int c = 0; c < CVC; ++c)
    Ams[lane * 64 + CSC + c] = (_Float16)((float)We[lane * 48 + 32 + c] * dts[c]);
#pragma unroll
  for (int k = OSS; k < 64; ++k) Ams[lane * 64 + k] = (_Float16)0.f;
  __syncthreads();

#pragma unroll
  for (int t = 0; t < 2; ++t) {
    v16h a0 = frag_a(Ams, t * 16, 0, 64);
    v16h a1 = frag_a(Ams, t * 16, 1, 64);
#pragma unroll
    for (int nt = 0; nt < 4; ++nt) {           // one 16-col tile == one head
      v8f c = {};
      c = wmma16(a0, frag_b_sw(sWaSw, 4, nt, 0), c);
      c = wmma16(a1, frag_b_sw(sWaSw, 4, nt, 1), c);
      const float adw = alpha_dot[nt * 16 + (lane & 15)];
      const int hh = lane >> 4;
#pragma unroll
      for (int r = 0; r < 8; ++r) {
        float vl = c[r];
        vl = vl > 0.f ? vl : 0.2f * vl;        // leaky_relu(0.2)
        float part = vl * adw;
        part += __shfl_xor(part, 1, 16);
        part += __shfl_xor(part, 2, 16);
        part += __shfl_xor(part, 4, 16);
        part += __shfl_xor(part, 8, 16);
        if ((lane & 15) == 0) {
          int row = t * 16 + r + hh * 8;
          logit[(size_t)(e0 + row) * HH + nt] = part;
          atomicMaxF(&mbuf[(size_t)Dst[row] * HH + nt], part);
        }
      }
    }
  }
}

// ---------------------------------------------------------------------------
// K3: ea = exp(logit - m[dst]); denom += ea
// ---------------------------------------------------------------------------
__global__ void softmax_kernel(const int* __restrict__ dstI,
                               const float* __restrict__ mbuf,
                               float* __restrict__ logit,
                               float* __restrict__ denom) {
  int i = blockIdx.x * blockDim.x + threadIdx.x;
  if (i >= NE * HH) return;
  int e = i >> 2, h = i & 3;
  int d = dstI[e];
  float ea = __expf(logit[i] - mbuf[d * HH + h]);
  logit[i] = ea;
  atomicAdd(&denom[d * HH + h], ea);
}

// ---------------------------------------------------------------------------
// K4: value path.  Per wave: 32 edges, 74 WMMA, scatter into agg.
// ---------------------------------------------------------------------------
__global__ void __launch_bounds__(64) value_kernel(
    const float* __restrict__ ssb, const float* __restrict__ dsb,
    const float* __restrict__ svb, const float* __restrict__ dvb,
    const float* __restrict__ rbf, const float* __restrict__ rsh,
    const float* __restrict__ W_rbf, const float* __restrict__ dtp2_w,
    const float* __restrict__ Wval_s, const float* __restrict__ Wval_v,
    const float* __restrict__ Wg, const float* __restrict__ Wvl_s,
    const float* __restrict__ Wvl_v,
    const int* __restrict__ srcI, const int* __restrict__ dstI,
    const float* __restrict__ eaBuf, const float* __restrict__ denom,
    float* __restrict__ agg) {
  __shared__ __align__(32) _Float16 sWvsSw[2 * 2 * 512];
  __shared__ __align__(32) _Float16 sWvvSw[2 * 1 * 512];
  __shared__ __align__(32) _Float16 sWlsSw[2 * 4 * 512];
  __shared__ __align__(32) _Float16 sWlvSw[2 * 2 * 512];
  __shared__ __align__(32) _Float16 sWr7[1 * 7 * 512];
  __shared__ float sWg[CSC * CVC];
  __shared__ float sDtp[WN];
  __shared__ __align__(32) _Float16 sBigA[2][32 * 64];    // rbf tile / ms_ / vs2
  __shared__ __align__(32) _Float16 sBigB[2][32 * WN];    // radial coeffs / mv_ tiles
  __shared__ __align__(32) _Float16 sVs[2][32 * 32];
  __shared__ __align__(32) _Float16 sVv[2][32 * 48];
  __shared__ float sAw[2][32 * HH];
  __shared__ int   sDstW[2][32];

  stage_b_swizzled(sWvsSw, Wval_s, OSS, 32, 2, 2);
  stage_b_swizzled(sWvvSw, Wval_v, 64, 16, 1, 2);
  stage_b_swizzled(sWlsSw, Wvl_s, OSS, 64, 4, 2);
  stage_b_swizzled(sWlvSw, Wvl_v, 64, 32, 2, 2);
  stage_b_swizzled(sWr7, W_rbf, NBANK, WN, 7, 1);
  for (int i = threadIdx.x; i < CSC * CVC; i += blockDim.x) sWg[i] = Wg[i];
  for (int i = threadIdx.x; i < WN; i += blockDim.x) sDtp[i] = dtp2_w[i];
  __syncthreads();

  const int wave = threadIdx.x >> 5;
  const int lane = threadIdx.x & 31;
  const int e0 = (blockIdx.x * 2 + wave) * 32;
  const int e  = e0 + lane;

  _Float16* RbfA = sBigA[wave];   // 32x32 halfs
  _Float16* Ams  = sBigA[wave];   // 32x64 halfs
  _Float16* We   = sBigB[wave];   // 32x112 halfs
  _Float16* Amv  = sBigB[wave];   // 32x64 halfs (after coeffs consumed)
  _Float16* Vs   = sVs[wave];
  _Float16* Vv   = sVv[wave];
  float*    Aw   = sAw[wave];
  int*      Dst  = sDstW[wave];

  const int sI = srcI[e], dI = dstI[e];
  Dst[lane] = dI;
#pragma unroll
  for (int h = 0; h < HH; ++h)
    Aw[lane * HH + h] = eaBuf[(size_t)e * HH + h] / denom[(size_t)dI * HH + h];

#pragma unroll
  for (int b = 0; b < NBANK; ++b)
    RbfA[lane * 32 + b] = (_Float16)rbf[(size_t)e * NBANK + b];
#pragma unroll
  for (int b = NBANK; b < 32; ++b) RbfA[lane * 32 + b] = (_Float16)0.f;

  const float y0 = rsh[(size_t)e * 4 + 0];
  float y1[3];
#pragma unroll
  for (int d = 0; d < 3; ++d) y1[d] = rsh[(size_t)e * 4 + 1 + d];

  float ms0[CSC];
#pragma unroll
  for (int c = 0; c < CSC; ++c)
    ms0[c] = ssb[(size_t)sI * CSC + c] + dsb[(size_t)dI * CSC + c];
  float mv0[CVC][3];
#pragma unroll
  for (int c = 0; c < CVC; ++c)
#pragma unroll
    for (int d = 0; d < 3; ++d)
      mv0[c][d] = svb[(size_t)sI * 48 + c * 3 + d] + dvb[(size_t)dI * 48 + c * 3 + d];
  __syncthreads();

  // radial weights for all 112 channels: rbf @ W_rbf  (14 WMMA)
#pragma unroll
  for (int t = 0; t < 2; ++t) {
    v16h a = frag_a(RbfA, t * 16, 0, 32);
#pragma unroll
    for (int nt = 0; nt < 7; ++nt) {
      v8f c = {};
      c = wmma16(a, frag_b_sw(sWr7, 7, nt, 0), c);
      const int n = nt * 16 + (lane & 15);
      const int hh = lane >> 4;
#pragma unroll
      for (int r = 0; r < 8; ++r)
        We[(t * 16 + r + hh * 8) * WN + n] = (_Float16)c[r];
    }
  }
  __syncthreads();

  // pull my edge's coefficients; build ms_ tile (overwrites rbf tile)
  float wsv[CSC], wvs[CVC], wvc[CVC];
#pragma unroll
  for (int c = 0; c < CSC; ++c) wsv[c] = (float)We[lane * WN + 32 + c];
#pragma unroll
  for (int c = 0; c < CVC; ++c) wvs[c] = (float)We[lane * WN + 64 + c];
#pragma unroll
  for (int c = 0; c < CVC; ++c) wvc[c] = (float)We[lane * WN + 96 + c];
#pragma unroll
  for (int c = 0; c < CSC; ++c)
    Ams[lane * 64 + c] = (_Float16)((float)We[lane * WN + c] * ms0[c] * y0);
#pragma unroll
  for (int c = 0; c < CVC; ++c) {
    float dt = (mv0[c][0] * y1[0] + mv0[c][1] * y1[1] + mv0[c][2] * y1[2]) * INV_SQRT3;
    Ams[lane * 64 + CSC + c] = (_Float16)((float)We[lane * WN + 80 + c] * dt);
  }
#pragma unroll
  for (int k = OSS; k < 64; ++k) Ams[lane * 64 + k] = (_Float16)0.f;
  __syncthreads();

  // vs = ms_ @ Wval_s  (8 WMMA)
#pragma unroll
  for (int t = 0; t < 2; ++t) {
    v16h a0 = frag_a(Ams, t * 16, 0, 64);
    v16h a1 = frag_a(Ams, t * 16, 1, 64);
#pragma unroll
    for (int nt = 0; nt < 2; ++nt) {
      v8f c = {};
      c = wmma16(a0, frag_b_sw(sWvsSw, 2, nt, 0), c);
      c = wmma16(a1, frag_b_sw(sWvsSw, 2, nt, 1), c);
      const int n = nt * 16 + (lane & 15);
      const int hh = lane >> 4;
#pragma unroll
      for (int r = 0; r < 8; ++r)
        Vs[(t * 16 + r + hh * 8) * 32 + n] = (_Float16)c[r];
    }
  }

  // vv[d] = mv_[d] @ Wval_v  (12 WMMA)
#pragma unroll
  for (int d = 0; d < 3; ++d) {
    __syncthreads();
#pragma unroll
    for (int c = 0; c < CSC; ++c)
      Amv[lane * 64 + c] = (_Float16)(wsv[c] * ms0[c] * y1[d]);
#pragma unroll
    for (int c = 0; c < CVC; ++c)
      Amv[lane * 64 + 32 + c] = (_Float16)(wvs[c] * mv0[c][d] * y0);
#pragma unroll
    for (int c = 0; c < CVC; ++c) {
      float cr;
      if (d == 0)      cr = mv0[c][1] * y1[2] - mv0[c][2] * y1[1];
      else if (d == 1) cr = mv0[c][2] * y1[0] - mv0[c][0] * y1[2];
      else             cr = mv0[c][0] * y1[1] - mv0[c][1] * y1[0];
      Amv[lane * 64 + 48 + c] = (_Float16)(wvc[c] * cr * INV_SQRT2);
    }
    __syncthreads();
#pragma unroll
    for (int t = 0; t < 2; ++t) {
      v16h a0 = frag_a(Amv, t * 16, 0, 64);
      v16h a1 = frag_a(Amv, t * 16, 1, 64);
      v8f c = {};
      c = wmma16(a0, frag_b_sw(sWvvSw, 1, 0, 0), c);
      c = wmma16(a1, frag_b_sw(sWvvSw, 1, 0, 1), c);
      const int ch = lane & 15;
      const int hh = lane >> 4;
#pragma unroll
      for (int r = 0; r < 8; ++r)
        Vv[(t * 16 + r + hh * 8) * 48 + ch * 3 + d] = (_Float16)c[r];
    }
  }
  __syncthreads();

  // gate + SiLU + second CG scalar part
  float vsv[CSC];
#pragma unroll
  for (int c = 0; c < CSC; ++c) vsv[c] = (float)Vs[lane * 32 + c];
  float gv[CVC];
#pragma unroll
  for (int g = 0; g < CVC; ++g) {
    float a = 0.f;
#pragma unroll
    for (int c = 0; c < CSC; ++c) a += vsv[c] * sWg[c * CVC + g];
    gv[g] = 1.f / (1.f + __expf(-a));
  }
#pragma unroll
  for (int c = 0; c < CSC; ++c) vsv[c] = vsv[c] / (1.f + __expf(-vsv[c]));
  float vvv[CVC][3];
#pragma unroll
  for (int c = 0; c < CVC; ++c)
#pragma unroll
    for (int d = 0; d < 3; ++d)
      vvv[c][d] = (float)Vv[lane * 48 + c * 3 + d] * gv[c];

#pragma unroll
  for (int c = 0; c < CSC; ++c)
    Ams[lane * 64 + c] = (_Float16)(sDtp[c] * vsv[c] * y0);
#pragma unroll
  for (int c = 0; c < CVC; ++c) {
    float dt = (vvv[c][0] * y1[0] + vvv[c][1] * y1[1] + vvv[c][2] * y1[2]) * INV_SQRT3;
    Ams[lane * 64 + 32 + c] = (_Float16)(sDtp[80 + c] * dt);
  }
#pragma unroll
  for (int k = OSS; k < 64; ++k) Ams[lane * 64 + k] = (_Float16)0.f;
  __syncthreads();

  // flat scalar half: vs2 @ Wvl_s -> weighted scatter  (16 WMMA)
#pragma unroll
  for (int t = 0; t < 2; ++t) {
    v16h a0 = frag_a(Ams, t * 16, 0, 64);
    v16h a1 = frag_a(Ams, t * 16, 1, 64);
#pragma unroll
    for (int nt = 0; nt < 4; ++nt) {
      v8f c = {};
      c = wmma16(a0, frag_b_sw(sWlsSw, 4, nt, 0), c);
      c = wmma16(a1, frag_b_sw(sWlsSw, 4, nt, 1), c);
      const int col = nt * 16 + (lane & 15);
      const int head = col / 40;
      const int hh = lane >> 4;
#pragma unroll
      for (int r = 0; r < 8; ++r) {
        int row = t * 16 + r + hh * 8;
        float w = Aw[row * HH + head];
        atomicAdd(&agg[(size_t)Dst[row] * FLATC + col], w * c[r]);
      }
    }
  }

  // flat vector half per component: vv2[d] @ Wvl_v -> weighted scatter (24 WMMA)
#pragma unroll
  for (int d = 0; d < 3; ++d) {
    __syncthreads();
#pragma unroll
    for (int c = 0; c < CSC; ++c)
      Amv[lane * 64 + c] = (_Float16)(sDtp[32 + c] * vsv[c] * y1[d]);
#pragma unroll
    for (int c = 0; c < CVC; ++c)
      Amv[lane * 64 + 32 + c] = (_Float16)(sDtp[64 + c] * vvv[c][d] * y0);
#pragma unroll
    for (int c = 0; c < CVC; ++c) {
      float cr;
      if (d == 0)      cr = vvv[c][1] * y1[2] - vvv[c][2] * y1[1];
      else if (d == 1) cr = vvv[c][2] * y1[0] - vvv[c][0] * y1[2];
      else             cr = vvv[c][0] * y1[1] - vvv[c][1] * y1[0];
      Amv[lane * 64 + 48 + c] = (_Float16)(sDtp[96 + c] * cr * INV_SQRT2);
    }
    __syncthreads();
#pragma unroll
    for (int t = 0; t < 2; ++t) {
      v16h a0 = frag_a(Amv, t * 16, 0, 64);
      v16h a1 = frag_a(Amv, t * 16, 1, 64);
#pragma unroll
      for (int nt = 0; nt < 2; ++nt) {
        v8f c = {};
        c = wmma16(a0, frag_b_sw(sWlvSw, 2, nt, 0), c);
        c = wmma16(a1, frag_b_sw(sWlvSw, 2, nt, 1), c);
        const int ch = nt * 16 + (lane & 15);
        const int fcol = 64 + ch * 3 + d;
        const int head = fcol / 40;
        const int hh = lane >> 4;
#pragma unroll
        for (int r = 0; r < 8; ++r) {
          int row = t * 16 + r + hh * 8;
          float w = Aw[row * HH + head];
          atomicAdd(&agg[(size_t)Dst[row] * FLATC + fcol], w * c[r]);
        }
      }
    }
  }
}

// ---------------------------------------------------------------------------
// K5: output projection + residual
// ---------------------------------------------------------------------------
__global__ void __launch_bounds__(256) out_kernel(
    const float* __restrict__ node, const float* __restrict__ agg,
    const float* __restrict__ Wp_s, const float* __restrict__ Wp_v,
    float* __restrict__ out) {
  __shared__ float sWp[64 * 32], sWpv[32 * 16];
  for (int i = threadIdx.x; i < 64 * 32; i += blockDim.x) sWp[i] = Wp_s[i];
  for (int i = threadIdx.x; i < 32 * 16; i += blockDim.x) sWpv[i] = Wp_v[i];
  __syncthreads();
  int n = blockIdx.x * blockDim.x + threadIdx.x;
  if (n >= NN) return;
  float fs[32];
#pragma unroll
  for (int j = 0; j < 32; ++j) fs[j] = 0.f;
  for (int k = 0; k < 64; ++k) {
    float av = agg[(size_t)n * FLATC + k];
#pragma unroll
    for (int j = 0; j < 32; ++j) fs[j] += av * sWp[k * 32 + j];
  }
  float fv[48];
#pragma unroll
  for (int i = 0; i < 48; ++i) fv[i] = 0.f;
  for (int c = 0; c < 32; ++c)
#pragma unroll
    for (int d = 0; d < 3; ++d) {
      float av = agg[(size_t)n * FLATC + 64 + c * 3 + d];
#pragma unroll
      for (int j = 0; j < 16; ++j) fv[j * 3 + d] += av * sWpv[c * 16 + j];
    }
#pragma unroll
  for (int j = 0; j < 32; ++j)
    out[(size_t)n * FEAT + j] = node[(size_t)n * FEAT + j] + fs[j];
#pragma unroll
  for (int i = 0; i < 48; ++i)
    out[(size_t)n * FEAT + 32 + i] = node[(size_t)n * FEAT + 32 + i] + fv[i];
}

// ---------------------------------------------------------------------------
extern "C" void kernel_launch(void* const* d_in, const int* in_sizes, int n_in,
                              void* d_out, int out_size, void* d_ws,
                              size_t ws_size, hipStream_t stream) {
  (void)in_sizes; (void)n_in; (void)out_size; (void)ws_size;
  const float* node      = (const float*)d_in[0];
  const float* rbf       = (const float*)d_in[1];
  const float* rsh       = (const float*)d_in[2];
  const float* Wsrc_s    = (const float*)d_in[3];
  const float* Wsrc_v    = (const float*)d_in[4];
  const float* Wdst_s    = (const float*)d_in[5];
  const float* Wdst_v    = (const float*)d_in[6];
  const float* W_rbf     = (const float*)d_in[7];
  const float* dtp2_w    = (const float*)d_in[8];
  const float* Wa        = (const float*)d_in[9];
  const float* alpha_dot = (const float*)d_in[10];
  const float* Wval_s    = (const float*)d_in[11];
  const float* Wval_v    = (const float*)d_in[12];
  const float* Wg        = (const float*)d_in[13];
  const float* Wvl_s     = (const float*)d_in[14];
  const float* Wvl_v     = (const float*)d_in[15];
  const float* Wp_s      = (const float*)d_in[16];
  const float* Wp_v      = (const float*)d_in[17];
  const int*   ei        = (const int*)d_in[18];
  const int* srcI = ei;
  const int* dstI = ei + NE;

  float* ws    = (float*)d_ws;
  float* ssb   = ws;
  float* dsb   = ssb + (size_t)NN * CSC;
  float* svb   = dsb + (size_t)NN * CSC;
  float* dvb   = svb + (size_t)NN * CVC * 3;
  float* logit = dvb + (size_t)NN * CVC * 3;
  float* mbuf  = logit + (size_t)NE * HH;
  float* denom = mbuf + (size_t)NN * HH;
  float* agg   = denom + (size_t)NN * HH;

  init_kernel<<<(NN * FLATC + 255) / 256, 256, 0, stream>>>(mbuf, denom, agg);
  node_kernel<<<(NN + 255) / 256, 256, 0, stream>>>(
      node, Wsrc_s, Wsrc_v, Wdst_s, Wdst_v, ssb, dsb, svb, dvb);
  attn_kernel<<<NE / 128, 128, 0, stream>>>(
      ssb, dsb, svb, dvb, rbf, rsh, W_rbf, Wa, alpha_dot, srcI, dstI, logit, mbuf);
  softmax_kernel<<<(NE * HH + 255) / 256, 256, 0, stream>>>(dstI, mbuf, logit, denom);
  value_kernel<<<NE / 64, 64, 0, stream>>>(
      ssb, dsb, svb, dvb, rbf, rsh, W_rbf, dtp2_w, Wval_s, Wval_v, Wg, Wvl_s,
      Wvl_v, srcI, dstI, logit, denom, agg);
  out_kernel<<<(NN + 255) / 256, 256, 0, stream>>>(node, agg, Wp_s, Wp_v,
                                                   (float*)d_out);
}
